// SolutionOfPDENet_26001732010600
// MI455X (gfx1250) — matrix operand order repaired and checked by reading, not compiled, not run on hardware
//
#include <hip/hip_runtime.h>
#include <hip/hip_bf16.h>

typedef __attribute__((ext_vector_type(16))) _Float16 v16h;
typedef __attribute__((ext_vector_type(8)))  _Float16 v8h;
typedef __attribute__((ext_vector_type(4)))  _Float16 v4h;
typedef __attribute__((ext_vector_type(8)))  float    v8f;

#define N_INT   128
#define HIDDEN  64
#define LATENT  8
#define IN_DIM  12
#define STAB    0.1f
#define MAX_ITER 6

// branch-free tanh: 1 - 2/(exp(2x)+1); exact limits at +/-inf via rcp(inf)=0
__device__ __forceinline__ float fast_tanh(float p) {
    float e2 = __expf(2.0f * p);                    // v_exp_f32 path, no branches
    return 1.0f - 2.0f * __builtin_amdgcn_rcpf(e2 + 1.0f);
}

__device__ __forceinline__ v16h cat8(v8h a, v8h b) {
    v16h r;
#pragma unroll
    for (int e = 0; e < 8; ++e) { r[e] = a[e]; r[e + 8] = b[e]; }
    return r;
}

__global__ __launch_bounds__(256)
void newton_pde_kernel(const float* __restrict__ u0,    // [B,128]
                       const float* __restrict__ zb,    // [B,8]
                       const float* __restrict__ xg,    // [130]
                       const float* __restrict__ W1,    // [12,64] row-major
                       const float* __restrict__ b1,    // [64]
                       const float* __restrict__ W2,    // [64]
                       const float* __restrict__ b2p,   // [1]
                       float* __restrict__ out)         // [B,128]
{
    const int b    = blockIdx.x;
    const int tid  = threadIdx.x;
    const int wave = tid >> 5;      // 0..7 : 16-row M-tile owned by this wave
    const int lane = tid & 31;
    const int nl   = lane & 15;     // column (B/C/D) or row (A) within tile
    const int hi   = lane >> 4;

    __shared__ float u_s[N_INT];
    __shared__ float uxx_s[N_INT];
    __shared__ float inv2h_s[N_INT];
    __shared__ float invh2_s[N_INT];
    __shared__ float P_s[N_INT];
    __shared__ float gu_s[N_INT], gux_s[N_INT], guxx_s[N_INT];
    __shared__ float dg_s[N_INT], upv_s[N_INT], lov_s[N_INT], rhs_s[N_INT];
    __shared__ float cp_s[N_INT], dp_s[N_INT], du_s[N_INT];
    __shared__ _Float16 feat_lds[N_INT * 32];       // padded features, A layout source
    __shared__ _Float16 h_lds[8 * 16 * HIDDEN];     // tanh activations
    __shared__ _Float16 s_lds[8 * 16 * HIDDEN];     // 1 - h^2

    // ---- one-time setup: state, geometry, invariant feature halves ------
    if (tid < N_INT) {
        u_s[tid] = u0[b * N_INT + tid];
        float xm = xg[tid], xc = xg[tid + 1], xp = xg[tid + 2];
        inv2h_s[tid] = 1.0f / (xp - xm);
        invh2_s[tid] = 1.0f / ((xp - xc) * (xc - xm));
        _Float16* fp = feat_lds + tid * 32;
#pragma unroll
        for (int k = 0; k < LATENT; ++k) fp[4 + k] = (_Float16)zb[b * LATENT + k];
#pragma unroll
        for (int k = IN_DIM; k < 32; ++k) fp[k] = (_Float16)0.0f;
    }

    // ---- weight fragments (batch & iteration invariant) ------------------
    v16h B1f[4];                                    // [32 x 64], K padded 12->32
#pragma unroll
    for (int t = 0; t < 4; ++t) {
#pragma unroll
        for (int e = 0; e < 16; ++e) {
            int k = (e & 7) + 8 * hi + 16 * (e >> 3); // B layout: K = (e&7) + 8*hi + 16*(e>>3)
            float v = (k < IN_DIM) ? W1[k * HIDDEN + t * 16 + nl] : 0.0f;
            B1f[t][e] = (_Float16)v;
        }
    }
    v16h B2f[2];                                    // [64 x 16] contraction weights
#pragma unroll
    for (int kk = 0; kk < 2; ++kk) {
#pragma unroll
        for (int e = 0; e < 16; ++e) {
            int k = (e & 7) + 8 * hi + 16 * (e >> 3) + 32 * kk;
            float w2 = W2[k];
            float v = 0.0f;
            if (nl == 0)      v = w2;
            else if (nl <= 3) v = w2 * W1[nl * HIDDEN + k];
            B2f[kk][e] = (_Float16)v;
        }
    }
    float b1v[4];
#pragma unroll
    for (int t = 0; t < 4; ++t) b1v[t] = b1[t * 16 + nl];
    const float b2v = b2p[0];
    const float xi  = xg[tid % N_INT + 1];          // only rows < 128 use it
    const int   m   = wave * 16 + nl;               // global row for A fragment

    // ---- Newton iterations ----------------------------------------------
    for (int it = 0; it < MAX_ITER; ++it) {
        __syncthreads();
        if (tid < N_INT) {
            float um = (tid > 0)         ? u_s[tid - 1] : 0.0f;
            float uc = u_s[tid];
            float up = (tid < N_INT - 1) ? u_s[tid + 1] : 0.0f;
            float ux  = (up - um) * inv2h_s[tid];
            float uxx = (up - 2.0f * uc + um) * invh2_s[tid];
            uxx_s[tid] = uxx;
            v4h f4;
            f4[0] = (_Float16)xi;  f4[1] = (_Float16)uc;
            f4[2] = (_Float16)ux;  f4[3] = (_Float16)uxx;
            *(v4h*)(feat_lds + tid * 32) = f4;      // one ds_store_b64
        }
        __syncthreads();

        // A fragment: two contiguous 16B blocks per lane (ds_load_b128 x2)
        const _Float16* fp = feat_lds + m * 32;
        v16h Af = cat8(*(const v8h*)(fp + 8 * hi),
                       *(const v8h*)(fp + 16 + 8 * hi));

        // GEMM1: pre = inp @ W1 + b1 ; h = tanh(pre) ; s = 1 - h^2
#pragma unroll
        for (int t = 0; t < 4; ++t) {
            v8f acc;
#pragma unroll
            for (int v = 0; v < 8; ++v) acc[v] = b1v[t];
            acc = __builtin_amdgcn_wmma_f32_16x16x32_f16(
                false, Af, false, B1f[t], (short)0, acc, false, false);
#pragma unroll
            for (int v = 0; v < 8; ++v) {
                float hh = fast_tanh(acc[v]);
                int row = v + 8 * hi;               // C/D layout row
                int idx = wave * 16 * HIDDEN + row * HIDDEN + t * 16 + nl;
                h_lds[idx] = (_Float16)hh;
                s_lds[idx] = (_Float16)(1.0f - hh * hh);
            }
        }
        __syncthreads();

        // Re-fragment h and s as A operands: 4 contiguous 16B loads each
        const _Float16* hp = h_lds + wave * 16 * HIDDEN + nl * HIDDEN;
        const _Float16* sp = s_lds + wave * 16 * HIDDEN + nl * HIDDEN;
        v16h Hf0 = cat8(*(const v8h*)(hp + 8 * hi), *(const v8h*)(hp + 16 + 8 * hi));
        v16h Hf1 = cat8(*(const v8h*)(hp + 32 + 8 * hi), *(const v8h*)(hp + 48 + 8 * hi));
        v16h Sf0 = cat8(*(const v8h*)(sp + 8 * hi), *(const v8h*)(sp + 16 + 8 * hi));
        v16h Sf1 = cat8(*(const v8h*)(sp + 32 + 8 * hi), *(const v8h*)(sp + 48 + 8 * hi));

        // GEMM2: P = h @ B2 (col0 = n_res), Q = s @ B2 (cols 1..3 = g_u,g_ux,g_uxx)
        v8f Pacc = {};
        Pacc = __builtin_amdgcn_wmma_f32_16x16x32_f16(
            false, Hf0, false, B2f[0], (short)0, Pacc, false, false);
        Pacc = __builtin_amdgcn_wmma_f32_16x16x32_f16(
            false, Hf1, false, B2f[1], (short)0, Pacc, false, false);
        v8f Qacc = {};
        Qacc = __builtin_amdgcn_wmma_f32_16x16x32_f16(
            false, Sf0, false, B2f[0], (short)0, Qacc, false, false);
        Qacc = __builtin_amdgcn_wmma_f32_16x16x32_f16(
            false, Sf1, false, B2f[1], (short)0, Qacc, false, false);

#pragma unroll
        for (int v = 0; v < 8; ++v) {
            int row = wave * 16 + v + 8 * hi;
            if      (nl == 0) P_s[row]    = Pacc[v];
            else if (nl == 1) gu_s[row]   = Qacc[v];
            else if (nl == 2) gux_s[row]  = Qacc[v];
            else if (nl == 3) guxx_s[row] = Qacc[v];
        }
        __syncthreads();

        // Tridiagonal Jacobian entries + residual
        if (tid < N_INT) {
            float gx  = guxx_s[tid] + STAB;
            float i2h = inv2h_s[tid], ih2 = invh2_s[tid];
            dg_s[tid]  = gu_s[tid] - 2.0f * ih2 * gx;
            upv_s[tid] = gux_s[tid] * i2h + ih2 * gx;
            lov_s[tid] = -gux_s[tid] * i2h + ih2 * gx;
            rhs_s[tid] = P_s[tid] + b2v + STAB * uxx_s[tid];
        }
        __syncthreads();

        // Thomas algorithm (serial, O(n))
        if (tid == 0) {
            cp_s[0] = upv_s[0] / dg_s[0];
            dp_s[0] = rhs_s[0] / dg_s[0];
            for (int i = 1; i < N_INT; ++i) {
                float minv = 1.0f / (dg_s[i] - lov_s[i] * cp_s[i - 1]);
                cp_s[i] = upv_s[i] * minv;
                dp_s[i] = (rhs_s[i] - lov_s[i] * dp_s[i - 1]) * minv;
            }
            du_s[N_INT - 1] = dp_s[N_INT - 1];
            for (int i = N_INT - 2; i >= 0; --i)
                du_s[i] = dp_s[i] - cp_s[i] * du_s[i + 1];
        }
        __syncthreads();
        if (tid < N_INT) u_s[tid] -= du_s[tid];     // DAMP = 1
    }

    __syncthreads();
    if (tid < N_INT) out[b * N_INT + tid] = u_s[tid];
}

extern "C" void kernel_launch(void* const* d_in, const int* in_sizes, int n_in,
                              void* d_out, int out_size, void* d_ws, size_t ws_size,
                              hipStream_t stream) {
    (void)n_in; (void)d_ws; (void)ws_size; (void)out_size;
    const float* u0  = (const float*)d_in[0];
    const float* zb  = (const float*)d_in[1];
    const float* xg  = (const float*)d_in[2];
    const float* W1  = (const float*)d_in[3];
    const float* b1  = (const float*)d_in[4];
    const float* W2  = (const float*)d_in[5];
    const float* b2  = (const float*)d_in[6];
    float* out = (float*)d_out;
    const int batch = in_sizes[0] / N_INT;   // 64
    newton_pde_kernel<<<batch, 256, 0, stream>>>(u0, zb, xg, W1, b1, W2, b2, out);
}
